// SpatialAttention_46909632807311
// MI455X (gfx1250) — matrix operand order repaired and checked by reading, not compiled, not run on hardware
//
#include <hip/hip_runtime.h>
#include <hip/hip_bf16.h>
#include <math.h>

// Problem constants (fixed by the reference)
#define BB   2
#define CCH  512        // channels
#define NN   4096       // H*W
#define GG   32         // groups
#define CPG  16         // channels per group
#define SCALE 0.04419417382415922f   // 1/sqrt(512)

typedef __bf16 bf16_t;
typedef __attribute__((ext_vector_type(16))) __bf16 v16bf;
typedef __attribute__((ext_vector_type(8)))  __bf16 v8bf;
typedef __attribute__((ext_vector_type(8)))  float  v8f;

// ---- workspace layout (bytes) ----
#define WQKV_OFF  0ull          // 1536*512 bf16            = 1,572,864
#define WPROJ_OFF 1572864ull    //  512*512 bf16            =   524,288
#define XN_OFF    2097152ull    // B*N*C bf16               = 8,388,608
#define QK_OFF    10485760ull   // B*N*1024 bf16 (q|k)      = 16,777,216
#define VT_OFF    27262976ull   // B*C*N bf16 (v transposed)= 8,388,608
#define ML_OFF    35651584ull   // 2*B*N fp32 (m,l)         =    65,536
#define PT_OFF    35717120ull   // B*N*N bf16               = 67,108,864
#define AO_OFF    XN_OFF        // attn-out reuses xn region (dead after QKV)

// ---- CDNA5 async global->LDS path (guarded; fallback = load + ds_store) ----
#if defined(__has_builtin)
#if __has_builtin(__builtin_amdgcn_global_load_async_to_lds_b128)
#define HAVE_ASYNC_LDS 1
#endif
#endif

#if defined(HAVE_ASYNC_LDS)
// Builtin expects: (v4i addrspace(1)* src, v4i addrspace(3)* dst, imm, imm)
typedef int v4i_t __attribute__((vector_size(16)));
typedef __attribute__((address_space(1))) v4i_t* g_v4i_p;   // 64-bit global ptr
typedef __attribute__((address_space(3))) v4i_t* l_v4i_p;   // 32-bit LDS offset
#endif

__device__ __forceinline__ void copy16_to_lds(const bf16_t* g, bf16_t* l) {
#if defined(HAVE_ASYNC_LDS)
  // Generic->AS1 is bit-identical on AMDGPU; generic shared ptr low 32 bits are
  // the LDS byte offset.
  g_v4i_p gp = __builtin_bit_cast(g_v4i_p, g);
  l_v4i_p lp = __builtin_bit_cast(l_v4i_p, (unsigned)(uintptr_t)l);
  __builtin_amdgcn_global_load_async_to_lds_b128(gp, lp, 0, 0);
#else
  *(v8bf*)l = *(const v8bf*)g;
#endif
}
__device__ __forceinline__ void wait_async_lds() {
#if defined(HAVE_ASYNC_LDS)
  asm volatile("s_wait_asynccnt 0x0" ::: "memory");
#endif
}

// ---------------- WMMA fragment helpers (wave32) ----------------
// A-matrix 16x32 bf16 fragment (ISA 7.12.2): lane m=l%16 holds row m;
// half h=l/16 selects K chunks [h*8,h*8+8) and [16+h*8,...).
__device__ __forceinline__ v16bf frag_a(const bf16_t* base, int stride) {
  int lane = threadIdx.x & 31;
  const bf16_t* row = base + (size_t)(lane & 15) * stride + ((lane >> 4) << 3);
  v16bf f;
  ((v8bf*)&f)[0] = *(const v8bf*)(row);
  ((v8bf*)&f)[1] = *(const v8bf*)(row + 16);
  return f;
}
// B-matrix 32x16 bf16 fragment: lane n=l%16 holds column n; half h=l/16 selects
// K chunks [h*8,h*8+8) and [16+h*8,...).  All B operands arranged so a "column" is a memory row.
__device__ __forceinline__ v16bf frag_b(const bf16_t* base, int stride) {
  int lane = threadIdx.x & 31;
  const bf16_t* row = base + (size_t)(lane & 15) * stride + ((lane >> 4) << 3);
  v16bf f;
  ((v8bf*)&f)[0] = *(const v8bf*)(row);
  ((v8bf*)&f)[1] = *(const v8bf*)(row + 16);
  return f;
}
__device__ __forceinline__ v8f wmma_bf16(v16bf a, v16bf b, v8f c) {
  return __builtin_amdgcn_wmma_f32_16x16x32_bf16(false, a, false, b, (short)0, c,
                                                 false, false);
}

// ---------------- K0: weights fp32 -> bf16 ----------------
__global__ void k_convert_w(const float* __restrict__ qkv_w,
                            const float* __restrict__ proj_w,
                            bf16_t* __restrict__ wq, bf16_t* __restrict__ wp) {
  int idx = blockIdx.x * 256 + threadIdx.x;
  const int nq = 3 * CCH * CCH;
  if (idx < nq)                 wq[idx] = (bf16_t)qkv_w[idx];
  else if (idx < nq + CCH*CCH)  wp[idx - nq] = (bf16_t)proj_w[idx - nq];
}

// ---------------- K1: GroupNorm, write transposed [n][c] bf16 ----------------
__global__ void k_groupnorm(const float* __restrict__ x,
                            const float* __restrict__ gw,
                            const float* __restrict__ gb,
                            bf16_t* __restrict__ xn) {
  int b = blockIdx.x / GG, g = blockIdx.x % GG;
  const float* xb = x + (size_t)b * CCH * NN + (size_t)g * CPG * NN;
  __shared__ float s1[256], s2[256], stats[2];
  float a = 0.f, q = 0.f;
  for (int idx = threadIdx.x; idx < CPG * NN; idx += 256) {
    float v = xb[idx]; a += v; q += v * v;
  }
  s1[threadIdx.x] = a; s2[threadIdx.x] = q;
  __syncthreads();
  for (int s = 128; s > 0; s >>= 1) {
    if ((int)threadIdx.x < s) {
      s1[threadIdx.x] += s1[threadIdx.x + s];
      s2[threadIdx.x] += s2[threadIdx.x + s];
    }
    __syncthreads();
  }
  if (threadIdx.x == 0) {
    float mu = s1[0] / (float)(CPG * NN);
    float var = s2[0] / (float)(CPG * NN) - mu * mu;
    stats[0] = mu; stats[1] = rsqrtf(var + 1e-6f);
  }
  __syncthreads();
  float mu = stats[0], rs = stats[1];
  for (int idx = threadIdx.x; idx < CPG * NN; idx += 256) {
    int ch = idx >> 12, n = idx & (NN - 1);
    int gc = g * CPG + ch;
    float v = (xb[idx] - mu) * rs * gw[gc] + gb[gc];
    xn[(size_t)b * NN * CCH + (size_t)n * CCH + gc] = (bf16_t)v;
  }
}

// ---------------- K2: QKV GEMM [4096x512]x[512x1536] ----------------
// q,k -> qk[b][n][0..1023]; v -> vT[b][c][n].  The 1024 split is aligned to the
// 128-wide block tile, so the store target is block-uniform (no exec juggling).
__global__ void k_gemm_qkv(const bf16_t* __restrict__ xn,
                           const bf16_t* __restrict__ wq,
                           const float* __restrict__ qkv_b,
                           bf16_t* __restrict__ qk, bf16_t* __restrict__ vT) {
  int b = blockIdx.z;
  int wave = threadIdx.x >> 5, lane = threadIdx.x & 31;
  int m0 = blockIdx.x * 128 + (wave & 1) * 64;
  int n0 = blockIdx.y * 128 + (wave >> 1) * 32;
  const bf16_t* A = xn + (size_t)b * NN * CCH;
  v8f acc[4][2] = {};
  for (int k = 0; k < CCH; k += 32) {
    v16bf af[4], bfr[2];
#pragma unroll
    for (int mt = 0; mt < 4; mt++)
      af[mt] = frag_a(A + (size_t)(m0 + mt * 16) * CCH + k, CCH);
#pragma unroll
    for (int nt = 0; nt < 2; nt++)
      bfr[nt] = frag_b(wq + (size_t)(n0 + nt * 16) * CCH + k, CCH);
#pragma unroll
    for (int mt = 0; mt < 4; mt++) {
#pragma unroll
      for (int nt = 0; nt < 2; nt++)
        acc[mt][nt] = wmma_bf16(af[mt], bfr[nt], acc[mt][nt]);
    }
  }
  int hi = lane >> 4, ln = lane & 15;
  if (blockIdx.y < 8) {           // q,k region (cols 0..1023), block-uniform
    bf16_t* qb = qk + (size_t)b * NN * 2 * CCH;
#pragma unroll
    for (int nt = 0; nt < 2; nt++) {
      int col = n0 + nt * 16 + ln;
      float bias = qkv_b[col];
#pragma unroll
      for (int mt = 0; mt < 4; mt++) {
#pragma unroll
        for (int v = 0; v < 8; v++) {
          int row = m0 + mt * 16 + v + hi * 8;
          qb[(size_t)row * 2 * CCH + col] = (bf16_t)(acc[mt][nt][v] + bias);
        }
      }
    }
  } else {                        // v region -> transposed [c][n]
    bf16_t* vb = vT + (size_t)b * CCH * NN;
#pragma unroll
    for (int nt = 0; nt < 2; nt++) {
      int col = n0 + nt * 16 + ln;
      float bias = qkv_b[col];
      int c = col - 2 * CCH;
#pragma unroll
      for (int mt = 0; mt < 4; mt++) {
#pragma unroll
        for (int v = 0; v < 8; v++) {
          int row = m0 + mt * 16 + v + hi * 8;
          vb[(size_t)c * NN + row] = (bf16_t)(acc[mt][nt][v] + bias);
        }
      }
    }
  }
}

// ---------------- K3: softmax row stats m[j], l[j] over scaled K^T Q ----------
__global__ void k_attn_stats(const bf16_t* __restrict__ qk,
                             float* __restrict__ mrow, float* __restrict__ lrow) {
  int b = blockIdx.y;
  int wave = threadIdx.x >> 5, lane = threadIdx.x & 31;
  int j0 = blockIdx.x * 128 + wave * 16;
  const bf16_t* qkb = qk + (size_t)b * NN * 2 * CCH;
  v16bf af[16];                      // hoisted K-row A-fragments (16 k-steps)
#pragma unroll
  for (int kk = 0; kk < 16; kk++)
    af[kk] = frag_a(qkb + (size_t)j0 * 2 * CCH + CCH + kk * 32, 2 * CCH);
  float pm[8], pl[8];
#pragma unroll
  for (int v = 0; v < 8; v++) { pm[v] = -1e30f; pl[v] = 0.f; }
  for (int i0 = 0; i0 < NN; i0 += 16) {
    v8f acc = {};
#pragma unroll
    for (int kk = 0; kk < 16; kk++) {
      v16bf bq = frag_b(qkb + (size_t)i0 * 2 * CCH + kk * 32, 2 * CCH);
      acc = wmma_bf16(af[kk], bq, acc);
    }
#pragma unroll
    for (int v = 0; v < 8; v++) {
      float s = acc[v] * SCALE;
      float nm = fmaxf(pm[v], s);
      pl[v] = pl[v] * __expf(pm[v] - nm) + __expf(s - nm);
      pm[v] = nm;
    }
  }
  __shared__ float sm[8][32][8], sl[8][32][8];
#pragma unroll
  for (int v = 0; v < 8; v++) { sm[wave][lane][v] = pm[v]; sl[wave][lane][v] = pl[v]; }
  __syncthreads();
  if (lane < 16) {
    int jj = lane, base = (jj < 8) ? 0 : 16, vg = jj & 7;
    float M = -1e30f, L = 0.f;
    for (int t = 0; t < 16; t++) {
      float mm = sm[wave][base + t][vg], ll = sl[wave][base + t][vg];
      float nm = fmaxf(M, mm);
      L = L * __expf(M - nm) + ll * __expf(mm - nm);
      M = nm;
    }
    mrow[b * NN + j0 + jj] = M;
    lrow[b * NN + j0 + jj] = L;
  }
}

// ---------------- K4a: PT[i][j] = exp(scale*(q_i . k_j) - m[j]) / l[j] --------
__global__ void k_attn_pt(const bf16_t* __restrict__ qk,
                          const float* __restrict__ mrow,
                          const float* __restrict__ lrow,
                          bf16_t* __restrict__ PT) {
  int b = blockIdx.z;
  int wave = threadIdx.x >> 5, lane = threadIdx.x & 31;
  int m0 = blockIdx.x * 128 + (wave & 1) * 64;   // i
  int n0 = blockIdx.y * 128 + (wave >> 1) * 32;  // j
  const bf16_t* qkb = qk + (size_t)b * NN * 2 * CCH;
  v8f acc[4][2] = {};
  for (int k = 0; k < CCH; k += 32) {
    v16bf af[4], bfr[2];
#pragma unroll
    for (int mt = 0; mt < 4; mt++)
      af[mt] = frag_a(qkb + (size_t)(m0 + mt * 16) * 2 * CCH + k, 2 * CCH);
#pragma unroll
    for (int nt = 0; nt < 2; nt++)
      bfr[nt] = frag_b(qkb + (size_t)(n0 + nt * 16) * 2 * CCH + CCH + k, 2 * CCH);
#pragma unroll
    for (int mt = 0; mt < 4; mt++) {
#pragma unroll
      for (int nt = 0; nt < 2; nt++)
        acc[mt][nt] = wmma_bf16(af[mt], bfr[nt], acc[mt][nt]);
    }
  }
  int hi = lane >> 4, ln = lane & 15;
#pragma unroll
  for (int nt = 0; nt < 2; nt++) {
    int j = n0 + nt * 16 + ln;
    float mj = mrow[b * NN + j];
    float rl = 1.0f / lrow[b * NN + j];
#pragma unroll
    for (int mt = 0; mt < 4; mt++) {
#pragma unroll
      for (int v = 0; v < 8; v++) {
        int i = m0 + mt * 16 + v + hi * 8;
        float p = __expf(acc[mt][nt][v] * SCALE - mj) * rl;
        PT[(size_t)b * NN * NN + (size_t)i * NN + j] = (bf16_t)p;
      }
    }
  }
}

// ---------------- K4b: attn out = PT[4096x4096] x V^T -> ao[n][c] ------------
// A (P^T) tile is staged into LDS via async global->LDS loads with double
// buffering; 4 waves share each A row so this cuts redundant global reads 4x.
__global__ void k_gemm_out(const bf16_t* __restrict__ PT,
                           const bf16_t* __restrict__ vT,
                           bf16_t* __restrict__ ao) {
  __shared__ bf16_t atile[2][128 * 32];
  int b = blockIdx.z;
  int wave = threadIdx.x >> 5, lane = threadIdx.x & 31;
  int mw = (wave & 1) * 64;                       // wave row offset in block tile
  int n0 = blockIdx.y * 128 + (wave >> 1) * 32;   // c
  const bf16_t* P = PT + (size_t)b * NN * NN + (size_t)(blockIdx.x * 128) * NN;
  const bf16_t* V = vT + (size_t)b * CCH * NN;
  v8f acc[4][2] = {};

  // stage tile k into buf: 128 rows x 32 cols bf16 = 512 x 16B, 2 chunks/thread
  auto stage = [&](int k, int buf) {
#pragma unroll
    for (int u = 0; u < 2; u++) {
      int c = (int)threadIdx.x + u * 256;         // chunk 0..511
      int row = c >> 2, k8 = (c & 3) * 8;
      copy16_to_lds(P + (size_t)row * NN + k + k8, &atile[buf][row * 32 + k8]);
    }
  };

  stage(0, 0);
  for (int k = 0; k < NN; k += 32) {
    int cur = (k >> 5) & 1;
    wait_async_lds();
    __syncthreads();                 // atile[cur] ready; all waves done with [cur^1]
    if (k + 32 < NN) stage(k + 32, cur ^ 1);
    v16bf af[4], bfr[2];
#pragma unroll
    for (int mt = 0; mt < 4; mt++)
      af[mt] = frag_a(&atile[cur][(mw + mt * 16) * 32], 32);
#pragma unroll
    for (int nt = 0; nt < 2; nt++)
      bfr[nt] = frag_b(V + (size_t)(n0 + nt * 16) * NN + k, NN);
#pragma unroll
    for (int mt = 0; mt < 4; mt++) {
#pragma unroll
      for (int nt = 0; nt < 2; nt++)
        acc[mt][nt] = wmma_bf16(af[mt], bfr[nt], acc[mt][nt]);
    }
  }
  int hi = lane >> 4, ln = lane & 15;
  int m0 = blockIdx.x * 128 + mw;
#pragma unroll
  for (int mt = 0; mt < 4; mt++) {
#pragma unroll
    for (int nt = 0; nt < 2; nt++) {
#pragma unroll
      for (int v = 0; v < 8; v++) {
        int i = m0 + mt * 16 + v + hi * 8;
        int c = n0 + nt * 16 + ln;
        ao[(size_t)b * NN * CCH + (size_t)i * CCH + c] = (bf16_t)acc[mt][nt][v];
      }
    }
  }
}

// ---------------- K5: proj GEMM + bias + residual -> [B][C][N] fp32 ----------
__global__ void k_gemm_proj(const bf16_t* __restrict__ ao,
                            const bf16_t* __restrict__ wp,
                            const float* __restrict__ proj_b,
                            const float* __restrict__ x,
                            float* __restrict__ out) {
  int b = blockIdx.z;
  int wave = threadIdx.x >> 5, lane = threadIdx.x & 31;
  int m0 = blockIdx.x * 128 + (wave & 1) * 64;   // i (position)
  int n0 = blockIdx.y * 128 + (wave >> 1) * 32;  // o (out channel)
  const bf16_t* A = ao + (size_t)b * NN * CCH;
  v8f acc[4][2] = {};
  for (int k = 0; k < CCH; k += 32) {
    v16bf af[4], bfr[2];
#pragma unroll
    for (int mt = 0; mt < 4; mt++)
      af[mt] = frag_a(A + (size_t)(m0 + mt * 16) * CCH + k, CCH);
#pragma unroll
    for (int nt = 0; nt < 2; nt++)
      bfr[nt] = frag_b(wp + (size_t)(n0 + nt * 16) * CCH + k, CCH);
#pragma unroll
    for (int mt = 0; mt < 4; mt++) {
#pragma unroll
      for (int nt = 0; nt < 2; nt++)
        acc[mt][nt] = wmma_bf16(af[mt], bfr[nt], acc[mt][nt]);
    }
  }
  int hi = lane >> 4, ln = lane & 15;
#pragma unroll
  for (int nt = 0; nt < 2; nt++) {
    int o = n0 + nt * 16 + ln;
    float bias = proj_b[o];
#pragma unroll
    for (int mt = 0; mt < 4; mt++) {
#pragma unroll
      for (int v = 0; v < 8; v++) {
        int i = m0 + mt * 16 + v + hi * 8;
        size_t idx = (size_t)b * CCH * NN + (size_t)o * NN + i;
        out[idx] = x[idx] + acc[mt][nt][v] + bias;
      }
    }
  }
}

extern "C" void kernel_launch(void* const* d_in, const int* in_sizes, int n_in,
                              void* d_out, int out_size, void* d_ws, size_t ws_size,
                              hipStream_t stream) {
  const float* x      = (const float*)d_in[0];
  const float* gn_w   = (const float*)d_in[1];
  const float* gn_b   = (const float*)d_in[2];
  const float* qkv_w  = (const float*)d_in[3];
  const float* qkv_b  = (const float*)d_in[4];
  const float* proj_w = (const float*)d_in[5];
  const float* proj_b = (const float*)d_in[6];
  char* ws = (char*)d_ws;
  bf16_t* wq   = (bf16_t*)(ws + WQKV_OFF);
  bf16_t* wp   = (bf16_t*)(ws + WPROJ_OFF);
  bf16_t* xn   = (bf16_t*)(ws + XN_OFF);
  bf16_t* qk   = (bf16_t*)(ws + QK_OFF);
  bf16_t* vT   = (bf16_t*)(ws + VT_OFF);
  float*  mrow = (float*)(ws + ML_OFF);
  float*  lrow = mrow + BB * NN;
  bf16_t* PT   = (bf16_t*)(ws + PT_OFF);
  bf16_t* ao   = (bf16_t*)(ws + AO_OFF);
  float*  out  = (float*)d_out;

  k_convert_w<<<(3 * CCH * CCH + CCH * CCH + 255) / 256, 256, 0, stream>>>(
      qkv_w, proj_w, wq, wp);
  k_groupnorm<<<BB * GG, 256, 0, stream>>>(x, gn_w, gn_b, xn);
  k_gemm_qkv<<<dim3(NN / 128, (3 * CCH) / 128, BB), 256, 0, stream>>>(
      xn, wq, qkv_b, qk, vT);
  k_attn_stats<<<dim3(NN / 128, BB), 256, 0, stream>>>(qk, mrow, lrow);
  k_attn_pt<<<dim3(NN / 128, NN / 128, BB), 256, 0, stream>>>(qk, mrow, lrow, PT);
  k_gemm_out<<<dim3(NN / 128, CCH / 128, BB), 256, 0, stream>>>(PT, vT, ao);
  k_gemm_proj<<<dim3(NN / 128, CCH / 128, BB), 256, 0, stream>>>(ao, wp, proj_b, x, out);
}
